// GraphTransformer_21449066676645
// MI455X (gfx1250) — matrix operand order repaired and checked by reading, not compiled, not run on hardware
//
#include <hip/hip_runtime.h>
#include <hip/hip_bf16.h>

// ---------------------------------------------------------------------------
// CDNA5 / gfx1250 Graph Transformer (3x TransformerConv + BN + ReLU)
// GEMMs use v_wmma_f32_16x16x32_bf16 (wave32, f32 accumulate).
// ---------------------------------------------------------------------------

typedef __attribute__((ext_vector_type(16))) __bf16 v16bf;
typedef __attribute__((ext_vector_type(8)))  float  v8f;

union FragBF16 { v16bf v; unsigned short u[16]; };

#define TILE_M 64
#define TILE_K 32
#define ON 128           // output channels, fixed for every GEMM in this model

// float -> bf16 (round to nearest even), bit-level (no scalar __bf16 math needed)
__device__ __forceinline__ unsigned short f2bf(float f) {
    unsigned int u = __float_as_uint(f);
    u += 0x7fffu + ((u >> 16) & 1u);
    return (unsigned short)(u >> 16);
}

__device__ __forceinline__ void atomicMaxFloat(float* addr, float val) {
    // works with init = -inf: positives via signed-int max, negatives via uint min
    if (val >= 0.0f) atomicMax((int*)addr, __float_as_int(val));
    else             atomicMin((unsigned int*)addr, __float_as_uint(val));
}

// ---------------------------------------------------------------------------
// Y[M,128] = X[M,K] * W[K,128] + bias   (K in {64,128}, multiple of 32)
// block = 256 threads = 8 waves; tile 64x128; wave grid 4(row) x 2(col)
// ---------------------------------------------------------------------------
__global__ __launch_bounds__(256) void gemm_bias_kernel(
    const float* __restrict__ X, const float* __restrict__ W,
    const float* __restrict__ bias, float* __restrict__ Y,
    int M, int K)
{
    __shared__ unsigned short As[TILE_M][TILE_K + 2];  // 64 x 34 bf16 (row-major)
    __shared__ unsigned short Bs[ON][TILE_K + 2];      // 128 x 34 bf16 (N-major)

    const int tid  = threadIdx.x;
    const int lane = tid & 31;
    const int wave = tid >> 5;
    const int wr   = wave >> 1;       // 0..3 : 16-row slab
    const int wc   = wave & 1;        // 0..1 : 64-col slab
    const int half = lane >> 4;       // lane group 0/1
    const int l15  = lane & 15;
    const int rowBase = blockIdx.x * TILE_M;

    v8f acc[4] = {};

    for (int k0 = 0; k0 < K; k0 += TILE_K) {
        // ---- stage X tile (64x32 f32 -> bf16), zero-fill OOB rows ----
        {
            const int r = tid >> 2;            // 0..63
            const int c = (tid & 3) * 8;       // 0,8,16,24
            const int grow = rowBase + r;
            float vals[8];
            if (grow < M) {
                const float* px = X + (size_t)grow * K + k0 + c;
#pragma unroll
                for (int i = 0; i < 8; ++i) vals[i] = px[i];
                if (k0 + TILE_K < K) __builtin_prefetch(px + TILE_K, 0, 1);
            } else {
#pragma unroll
                for (int i = 0; i < 8; ++i) vals[i] = 0.0f;
            }
#pragma unroll
            for (int i = 0; i < 8; ++i) As[r][c + i] = f2bf(vals[i]);
        }
        // ---- stage W tile (32x128 f32) transposed -> Bs[n][k] bf16 ----
        {
            const int kk = tid >> 3;           // 0..31
            const int ng = (tid & 7) * 16;     // 0..112
            const float* pw = W + (size_t)(k0 + kk) * ON + ng;
#pragma unroll
            for (int i = 0; i < 16; ++i) Bs[ng + i][kk] = f2bf(pw[i]);
        }
        __syncthreads();

        // ---- A fragment: 16x32 bf16, ISA 7.12.2 layout ----
        FragBF16 a;
        const int arow = wr * 16 + l15;
#pragma unroll
        for (int vi = 0; vi < 8; ++vi) {
            const int klo = ((vi < 4) ? 2 * vi : 2 * vi + 8) + (half ? 8 : 0);
            a.u[2 * vi]     = As[arow][klo];
            a.u[2 * vi + 1] = As[arow][klo + 1];
        }
        // ---- 4 B fragments (32x16 each) + WMMA ----
#pragma unroll
        for (int t = 0; t < 4; ++t) {
            FragBF16 b;
            const int n = wc * 64 + t * 16 + l15;
#pragma unroll
            for (int vi = 0; vi < 8; ++vi) {
                const int k = ((vi < 4) ? 2 * vi : 2 * vi + 8) + (half ? 8 : 0);
                b.u[2 * vi]     = Bs[n][k];
                b.u[2 * vi + 1] = Bs[n][k + 1];
            }
            acc[t] = __builtin_amdgcn_wmma_f32_16x16x32_bf16(
                false, a.v, false, b.v, (short)0, acc[t], false, false);
        }
        __syncthreads();
    }

    // ---- epilogue: bias + store (C layout: VGPR j -> M = j + half*8) ----
#pragma unroll
    for (int t = 0; t < 4; ++t) {
        const int col = wc * 64 + t * 16 + l15;
        const float bv = bias[col];
#pragma unroll
        for (int j = 0; j < 8; ++j) {
            const int row = rowBase + wr * 16 + (half ? 8 : 0) + j;
            if (row < M) Y[(size_t)row * ON + col] = acc[t][j] + bv;
        }
    }
}

// ---------------------------------------------------------------------------
// attention passes
// ---------------------------------------------------------------------------
__global__ void init_ms_kernel(float* m, float* s, int n) {
    const int i = blockIdx.x * blockDim.x + threadIdx.x;
    if (i < n) { m[i] = __int_as_float(0xff800000); s[i] = 0.0f; }
}

// one thread per (edge, head): alpha = <q[dst], k[src]+e> / sqrt(32); atomic max
__global__ __launch_bounds__(256) void edge_alpha_kernel(
    const float* __restrict__ q, const float* __restrict__ k,
    const float* __restrict__ e, const long long* __restrict__ src,
    const long long* __restrict__ dst, float* __restrict__ alpha,
    float* __restrict__ m, int Ecnt)
{
    const int idx = blockIdx.x * blockDim.x + threadIdx.x;
    if (idx >= Ecnt * 4) return;
    const int eid = idx >> 2, h = idx & 3;
    const int s = (int)src[eid], d = (int)dst[eid];
    const float* qp = q + (size_t)d * 128 + h * 32;
    const float* kp = k + (size_t)s * 128 + h * 32;
    const float* ep = e + (size_t)eid * 128 + h * 32;
    float acc = 0.0f;
#pragma unroll
    for (int i = 0; i < 32; ++i) acc = fmaf(qp[i], kp[i] + ep[i], acc);
    acc *= 0.17677669529663687f;   // 1/sqrt(32)
    alpha[idx] = acc;
    atomicMaxFloat(&m[d * 4 + h], acc);
}

__global__ __launch_bounds__(256) void edge_exp_kernel(
    const float* __restrict__ m, const long long* __restrict__ dst,
    float* __restrict__ alpha, float* __restrict__ ssum, int Ecnt)
{
    const int idx = blockIdx.x * blockDim.x + threadIdx.x;
    if (idx >= Ecnt * 4) return;
    const int eid = idx >> 2, h = idx & 3;
    const int d = (int)dst[eid];
    const float a = __expf(alpha[idx] - m[d * 4 + h]);
    alpha[idx] = a;
    atomicAdd(&ssum[d * 4 + h], a);
}

// 2 edges per 256-thread block; out[dst] += coef * (v[src] + e)
__global__ __launch_bounds__(256) void edge_aggregate_kernel(
    const float* __restrict__ v, const float* __restrict__ e,
    const float* __restrict__ a, const float* __restrict__ ssum,
    const long long* __restrict__ src, const long long* __restrict__ dst,
    float* __restrict__ out, int Ecnt)
{
    const int eid = blockIdx.x * 2 + (threadIdx.x >> 7);
    if (eid >= Ecnt) return;
    const int c = threadIdx.x & 127;
    const int h = c >> 5;
    const int s = (int)src[eid], d = (int)dst[eid];
    const float coef = a[eid * 4 + h] / (ssum[d * 4 + h] + 1e-16f);
    const float val = coef * (v[(size_t)s * 128 + c] + e[(size_t)eid * 128 + c]);
    atomicAdd(&out[(size_t)d * 128 + c], val);
}

// ---------------------------------------------------------------------------
// BatchNorm1d (population stats) + ReLU
// ---------------------------------------------------------------------------
__global__ void bn_stats_init_kernel(float* sum, float* sumsq) {
    sum[threadIdx.x] = 0.0f; sumsq[threadIdx.x] = 0.0f;
}

__global__ __launch_bounds__(256) void bn_stats_kernel(
    const float* __restrict__ h, float* __restrict__ sum,
    float* __restrict__ sumsq, int n)
{
    const int c = threadIdx.x & 127;
    const int r0 = blockIdx.x * 2 + (threadIdx.x >> 7);
    const int rstep = gridDim.x * 2;
    float s = 0.0f, s2 = 0.0f;
    for (int r = r0; r < n; r += rstep) {
        const float x = h[(size_t)r * 128 + c];
        s += x; s2 = fmaf(x, x, s2);
    }
    atomicAdd(&sum[c], s);
    atomicAdd(&sumsq[c], s2);
}

__global__ __launch_bounds__(256) void bn_apply_kernel(
    float* __restrict__ h, const float* __restrict__ sum,
    const float* __restrict__ sumsq, const float* __restrict__ gamma,
    const float* __restrict__ beta, int n)
{
    const int idx = blockIdx.x * blockDim.x + threadIdx.x;
    if (idx >= n * 128) return;
    const int c = idx & 127;
    const float inv_n = 1.0f / (float)n;
    const float mu  = sum[c] * inv_n;
    const float var = sumsq[c] * inv_n - mu * mu;
    const float y = (h[idx] - mu) * __frsqrt_rn(var + 1e-5f) * gamma[c] + beta[c];
    h[idx] = fmaxf(y, 0.0f);
}

// ---------------------------------------------------------------------------
// host-side orchestration
// ---------------------------------------------------------------------------
extern "C" void kernel_launch(void* const* d_in, const int* in_sizes, int n_in,
                              void* d_out, int out_size, void* d_ws, size_t ws_size,
                              hipStream_t stream) {
    (void)n_in; (void)out_size; (void)ws_size;

    const float*     x         = (const float*)d_in[0];
    const long long* adj       = (const long long*)d_in[1];   // int64 per reference
    const float*     edge_attr = (const float*)d_in[2];
    const int N = in_sizes[0] / 64;
    const int E = in_sizes[2] / 64;
    const long long* src = adj;
    const long long* dst = adj + E;

    // workspace carve-up (floats)
    float* ws = (float*)d_ws;
    size_t o = 0;
    float* qb = ws + o;  o += (size_t)N * 128;
    float* kb = ws + o;  o += (size_t)N * 128;
    float* vb = ws + o;  o += (size_t)N * 128;
    float* hA = ws + o;  o += (size_t)N * 128;
    float* hB = ws + o;  o += (size_t)N * 128;
    float* eb = ws + o;  o += (size_t)E * 128;
    float* ab = ws + o;  o += (size_t)E * 4;
    float* mb = ws + o;  o += (size_t)N * 4;
    float* sb = ws + o;  o += (size_t)N * 4;
    float* bnsum = ws + o; o += 128;
    float* bnss  = ws + o; o += 128;

    const int gN = (N + TILE_M - 1) / TILE_M;
    const int gE = (E + TILE_M - 1) / TILE_M;
    const int gEH = (E * 4 + 255) / 256;

    auto layer = [&](const float* X, int K, int pi, float* Y,
                     const float* gamma, const float* beta) {
        const float* Wq = (const float*)d_in[pi + 0]; const float* bq = (const float*)d_in[pi + 1];
        const float* Wk = (const float*)d_in[pi + 2]; const float* bk = (const float*)d_in[pi + 3];
        const float* Wv = (const float*)d_in[pi + 4]; const float* bv = (const float*)d_in[pi + 5];
        const float* We = (const float*)d_in[pi + 6]; const float* be = (const float*)d_in[pi + 7];
        const float* Ws = (const float*)d_in[pi + 8]; const float* bs = (const float*)d_in[pi + 9];

        gemm_bias_kernel<<<gN, 256, 0, stream>>>(X, Wq, bq, qb, N, K);
        gemm_bias_kernel<<<gN, 256, 0, stream>>>(X, Wk, bk, kb, N, K);
        gemm_bias_kernel<<<gN, 256, 0, stream>>>(X, Wv, bv, vb, N, K);
        gemm_bias_kernel<<<gE, 256, 0, stream>>>(edge_attr, We, be, eb, E, 64);
        gemm_bias_kernel<<<gN, 256, 0, stream>>>(X, Ws, bs, Y, N, K);  // skip term seeds Y

        init_ms_kernel<<<(N * 4 + 255) / 256, 256, 0, stream>>>(mb, sb, N * 4);
        edge_alpha_kernel<<<gEH, 256, 0, stream>>>(qb, kb, eb, src, dst, ab, mb, E);
        edge_exp_kernel<<<gEH, 256, 0, stream>>>(mb, dst, ab, sb, E);
        edge_aggregate_kernel<<<(E + 1) / 2, 256, 0, stream>>>(vb, eb, ab, sb, src, dst, Y, E);

        if (gamma) {
            bn_stats_init_kernel<<<1, 128, 0, stream>>>(bnsum, bnss);
            bn_stats_kernel<<<256, 256, 0, stream>>>(Y, bnsum, bnss, N);
            bn_apply_kernel<<<(N * 128 + 255) / 256, 256, 0, stream>>>(Y, bnsum, bnss, gamma, beta, N);
        }
    };

    // d_in layout: 0 x, 1 adj, 2 edge_attr, then per layer
    // {Wq,bq,Wk,bk,Wv,bv,We,be,Ws,bs[,gamma,beta]}
    layer(x,  64,  3, hA, (const float*)d_in[13], (const float*)d_in[14]);
    layer(hA, 128, 15, hB, (const float*)d_in[25], (const float*)d_in[26]);
    layer(hB, 128, 27, (float*)d_out, nullptr, nullptr);

    // second tuple output: edge_attr passthrough
    hipMemcpyAsync((float*)d_out + (size_t)N * 128, edge_attr,
                   (size_t)E * 64 * sizeof(float),
                   hipMemcpyDeviceToDevice, stream);
}
